// RankOnePools_38835094290478
// MI455X (gfx1250) — matrix operand order repaired and checked by reading, not compiled, not run on hardware
//
#include <hip/hip_runtime.h>

// RankOnePools on MI455X (gfx1250): densify the rank-one gather into two
// bf16 WMMA GEMMs + per-token LDS histogram.
//   P   = X @ SVH^T                      [T,EK]   v_wmma_f32_16x16x32_bf16
//   W   = cnt(index) * P  -> bf16        [T,EK]   LDS ds-atomics histogram
//   OUT = W @ U^T                        [T,D_OUT]v_wmma_f32_16x16x32_bf16
// EK=256 is tiny, so the 8x FLOP expansion is free: bf16 matrix math puts us
// on the ~3us HBM roofline (72MB compulsory traffic @ 23.3 TB/s).
// B panels are staged into LDS with the Tensor Data Mover (tensor_load_to_lds
// + s_wait_tensorcnt) so WMMAs read B via ds_load instead of stalling on L2.

#define T_TOK 4096
#define S_SEL 32
#define DIN   2048
#define DOUT  2048
#define EKN   256

typedef __attribute__((ext_vector_type(16))) __bf16 v16bf;
typedef __attribute__((ext_vector_type(8)))  __bf16 v8bf;
typedef __attribute__((ext_vector_type(8)))  float  v8f;
typedef __attribute__((ext_vector_type(4)))  unsigned int v4u;
typedef __attribute__((ext_vector_type(8)))  int v8i_t;
typedef __attribute__((ext_vector_type(4)))  int v4i_t;

#define HAVE_TDM __has_builtin(__builtin_amdgcn_tensor_load_to_lds)

// Byte offset of an LDS object within the workgroup allocation: generic LDS
// addresses carry the offset in the low 32 bits (aperture in the high bits).
__device__ __forceinline__ unsigned lds_off(const void* p) {
  return (unsigned)(unsigned long long)(uintptr_t)p;
}

#if HAVE_TDM
// Issue one TDM 2D-tile load: rows x width bf16 elements, row stride
// row_stride_elems, packed row-major (width*2 bytes per row) into LDS.
// Descriptor packing per CDNA5 ISA ch.10.8 (D# groups 0/1; groups 2/3 zero
// for 2D tiles).
__device__ __forceinline__ void tdm_load_2d_bf16(
    unsigned lds_byte_off, const void* gptr,
    unsigned width_elems, unsigned rows, unsigned row_stride_elems) {
  const unsigned long long ga = (unsigned long long)(uintptr_t)gptr;
  v4u g0;
  g0[0] = 1u;                                   // count=1, user descriptor
  g0[1] = lds_byte_off;                         // lds_addr[31:0]
  g0[2] = (unsigned)ga;                         // global_addr[31:0]
  g0[3] = (unsigned)((ga >> 32) & 0x1FFFFFFu)   // global_addr[56:32]
          | (2u << 30);                         // type = 2 ("image")
  v8i_t g1;
  g1[0] = (int)(1u << 16);                      // data_size=1 -> 2 bytes
  g1[1] = (int)((width_elems & 0xFFFFu) << 16); // tensor_dim0[15:0]
  g1[2] = (int)(((width_elems >> 16) & 0xFFFFu) |
                ((rows & 0xFFFFu) << 16));      // td0 hi | tensor_dim1 lo
  g1[3] = (int)(((rows >> 16) & 0xFFFFu) |
                ((width_elems & 0xFFFFu) << 16)); // td1 hi | tile_dim0
  g1[4] = (int)(rows & 0xFFFFu);                // tile_dim1 | tile_dim2=0
  g1[5] = (int)row_stride_elems;                // tensor_dim0_stride[31:0]
  g1[6] = 0;                                    // stride0 hi | stride1 lo
  g1[7] = 0;
  const v4i_t gz = {0, 0, 0, 0};
#if defined(__clang_major__) && __clang_major__ >= 23
  const v8i_t gz8 = {0, 0, 0, 0, 0, 0, 0, 0};
  __builtin_amdgcn_tensor_load_to_lds(g0, g1, gz, gz, gz8, 0);
#else
  __builtin_amdgcn_tensor_load_to_lds(g0, g1, gz, gz, 0);
#endif
}
#endif  // HAVE_TDM

// ---------------------------------------------------------------------------
// Kernel 1: build bf16 transposed operand matrices.
//   svhT[i][e] = svh[e][i]   (GEMM1 B-rows over K=i contiguous in e)
//   utb [e][o] = u[o][e]     (GEMM2 B-rows over K=e contiguous in o)
// ---------------------------------------------------------------------------
__global__ __launch_bounds__(256) void prep_transpose(
    const float* __restrict__ u, const float* __restrict__ svh,
    __bf16* __restrict__ svhT, __bf16* __restrict__ utb) {
  const int idx = blockIdx.x * 256 + threadIdx.x;   // 0 .. DIN*EKN-1
  const int e = idx & (EKN - 1);
  const int i = idx >> 8;                           // EKN == 256
  svhT[(size_t)i * EKN + e]  = (__bf16)svh[(size_t)e * DIN + i];
  utb [(size_t)e * DOUT + i] = (__bf16)u[(size_t)i * EKN + e];
}

// ---------------------------------------------------------------------------
// WMMA fragment layouts (wave32, 16x16x32 bf16 -> f32):
// A 16x32: lane row M=lane&15; elems 0..7 = K[h*8..+7], 8..15 = K[16+h*8..+7]
//          with h = lane>>4.
// B 32x16: lane col N=lane&15; elems 0..15 over K in the same order as A.
// C/D:     col N = lane&15, elem r = row M = r + 8*(lane>>4).
// ---------------------------------------------------------------------------

// GEMM1:  P[T,EKN] = X[T,DIN](f32 -> bf16 on the fly) @ svhT
// B staged in LDS via double-buffered 32x64 TDM chunks.
__global__ __launch_bounds__(256) void gemm1_x_svhT(
    const float* __restrict__ x, const __bf16* __restrict__ svhT,
    float* __restrict__ P) {
  __shared__ __attribute__((aligned(128))) __bf16 bpanel[2][32 * 64];  // 2x4KB
  const int lane = threadIdx.x & 31;
  const int wave = threadIdx.x >> 5;
  const int half = lane >> 4;
  const int mrow = lane & 15;
  const int mBase = blockIdx.x * 128 + wave * 16;   // 8 waves stacked in M
  const int nBase = blockIdx.y * 64;                // 4 N-tiles per wave

  const float* arow  = x + (size_t)(mBase + mrow) * DIN;
  const __bf16* bsrc = svhT + nBase;                // row k at bsrc + k*EKN

#if HAVE_TDM
  const bool issuer = (threadIdx.x < 32);           // wave 0 drives the TDM
  if (issuer) tdm_load_2d_bf16(lds_off(&bpanel[0][0]), bsrc, 64, 32, EKN);
#endif

  v8f acc0 = {}, acc1 = {}, acc2 = {}, acc3 = {};

  for (int k0 = 0; k0 < DIN; k0 += 32) {
    const int cur = (k0 >> 5) & 1;
#if HAVE_TDM
    if (issuer) {
      if (k0 + 32 < DIN) {
        tdm_load_2d_bf16(lds_off(&bpanel[cur ^ 1][0]),
                         bsrc + (size_t)(k0 + 32) * EKN, 64, 32, EKN);
        __builtin_amdgcn_s_wait_tensorcnt(1);       // chunk k0 has landed
      } else {
        __builtin_amdgcn_s_wait_tensorcnt(0);
      }
    }
    __syncthreads();                                // LDS chunk visible to all
#else
    __syncthreads();
    {   // fallback staging: 256 threads x v8bf = 4KB chunk
      const int r = threadIdx.x >> 3, c = (threadIdx.x & 7) * 8;
      *(v8bf*)&bpanel[0][r * 64 + c] =
          *(const v8bf*)(bsrc + (size_t)(k0 + r) * EKN + c);
    }
    __syncthreads();
#endif

    if (k0 + 32 < DIN) __builtin_prefetch(arow + k0 + 32 + half * 8, 0, 3);

    // A fragment: two 32B f32 chunks of row (mBase+mrow), converted to bf16.
    const float4 f0 = *(const float4*)(arow + k0 + half * 8);
    const float4 f1 = *(const float4*)(arow + k0 + half * 8 + 4);
    const float4 f2 = *(const float4*)(arow + k0 + 16 + half * 8);
    const float4 f3 = *(const float4*)(arow + k0 + 16 + half * 8 + 4);
    v16bf a;
    a[0]  = (__bf16)f0.x; a[1]  = (__bf16)f0.y; a[2]  = (__bf16)f0.z; a[3]  = (__bf16)f0.w;
    a[4]  = (__bf16)f1.x; a[5]  = (__bf16)f1.y; a[6]  = (__bf16)f1.z; a[7]  = (__bf16)f1.w;
    a[8]  = (__bf16)f2.x; a[9]  = (__bf16)f2.y; a[10] = (__bf16)f2.z; a[11] = (__bf16)f2.w;
    a[12] = (__bf16)f3.x; a[13] = (__bf16)f3.y; a[14] = (__bf16)f3.z; a[15] = (__bf16)f3.w;

    // B fragments from LDS: lane holds column N=lane&15, elems over K in A's order.
#if HAVE_TDM
    const __bf16* brow = &bpanel[cur][mrow];
#else
    const __bf16* brow = &bpanel[0][mrow];
    (void)cur;
#endif
    v16bf b0, b1, b2, b3;
#pragma unroll
    for (int s = 0; s < 16; ++s) {
      const int kk = (s < 8) ? (half * 8 + s) : (16 + half * 8 + (s - 8));
      b0[s] = brow[kk * 64];
      b1[s] = brow[kk * 64 + 16];
      b2[s] = brow[kk * 64 + 32];
      b3[s] = brow[kk * 64 + 48];
    }

    acc0 = __builtin_amdgcn_wmma_f32_16x16x32_bf16(false, a, false, b0, (short)0, acc0, false, false);
    acc1 = __builtin_amdgcn_wmma_f32_16x16x32_bf16(false, a, false, b1, (short)0, acc1, false, false);
    acc2 = __builtin_amdgcn_wmma_f32_16x16x32_bf16(false, a, false, b2, (short)0, acc2, false, false);
    acc3 = __builtin_amdgcn_wmma_f32_16x16x32_bf16(false, a, false, b3, (short)0, acc3, false, false);

#if HAVE_TDM
    __syncthreads();   // everyone done with bpanel[cur] before TDM reuses it
#endif
  }

  float* pout = P + (size_t)(mBase + 8 * half) * EKN + nBase + mrow;
#pragma unroll
  for (int r = 0; r < 8; ++r) {
    pout[(size_t)r * EKN + 0]  = acc0[r];
    pout[(size_t)r * EKN + 16] = acc1[r];
    pout[(size_t)r * EKN + 32] = acc2[r];
    pout[(size_t)r * EKN + 48] = acc3[r];
  }
}

// ---------------------------------------------------------------------------
// Kernel 3: per-token histogram of index (LDS atomics), W = cnt * P -> bf16.
// ---------------------------------------------------------------------------
__global__ __launch_bounds__(256) void build_w(
    const float* __restrict__ P, const int* __restrict__ index,
    __bf16* __restrict__ Wbf) {
  __shared__ int cnt[EKN];
  const int t = blockIdx.x;
  const int e = threadIdx.x;
  cnt[e] = 0;
  __syncthreads();
  if (e < S_SEL) atomicAdd(&cnt[index[(size_t)t * S_SEL + e] & (EKN - 1)], 1);
  __syncthreads();
  Wbf[(size_t)t * EKN + e] = (__bf16)((float)cnt[e] * P[(size_t)t * EKN + e]);
}

// GEMM2:  OUT[T,DOUT] = W[T,EKN](bf16) @ utb
// Full B panel (256 x 64 = 32KB) staged once per block via one TDM load.
__global__ __launch_bounds__(256) void gemm2_w_ut(
    const __bf16* __restrict__ Wbf, const __bf16* __restrict__ utb,
    float* __restrict__ out) {
  __shared__ __attribute__((aligned(128))) __bf16 bpanel[EKN * 64];  // 32KB
  const int lane = threadIdx.x & 31;
  const int wave = threadIdx.x >> 5;
  const int half = lane >> 4;
  const int mrow = lane & 15;
  const int mBase = blockIdx.x * 128 + wave * 16;
  const int nBase = blockIdx.y * 64;

#if HAVE_TDM
  if (threadIdx.x < 32) {
    tdm_load_2d_bf16(lds_off(&bpanel[0]), utb + nBase, 64, EKN, DOUT);
    __builtin_amdgcn_s_wait_tensorcnt(0);
  }
#else
  for (int i = threadIdx.x * 8; i < EKN * 64; i += 256 * 8) {
    const int r = i >> 6, c = i & 63;
    *(v8bf*)&bpanel[i] = *(const v8bf*)(utb + (size_t)r * DOUT + nBase + c);
  }
#endif
  __syncthreads();

  const __bf16* arow = Wbf + (size_t)(mBase + mrow) * EKN;

  v8f acc0 = {}, acc1 = {}, acc2 = {}, acc3 = {};

#pragma unroll
  for (int k0 = 0; k0 < EKN; k0 += 32) {
    // A fragment: two 16B bf16 chunks of row (mBase+mrow).
    const v8bf lo = *(const v8bf*)(arow + k0 + half * 8);
    const v8bf hi = *(const v8bf*)(arow + k0 + 16 + half * 8);
    v16bf a;
#pragma unroll
    for (int i = 0; i < 8; ++i) { a[i] = lo[i]; a[i + 8] = hi[i]; }

    const __bf16* brow = &bpanel[k0 * 64 + mrow];
    v16bf b0, b1, b2, b3;
#pragma unroll
    for (int s = 0; s < 16; ++s) {
      const int kk = (s < 8) ? (half * 8 + s) : (16 + half * 8 + (s - 8));
      b0[s] = brow[kk * 64];
      b1[s] = brow[kk * 64 + 16];
      b2[s] = brow[kk * 64 + 32];
      b3[s] = brow[kk * 64 + 48];
    }

    acc0 = __builtin_amdgcn_wmma_f32_16x16x32_bf16(false, a, false, b0, (short)0, acc0, false, false);
    acc1 = __builtin_amdgcn_wmma_f32_16x16x32_bf16(false, a, false, b1, (short)0, acc1, false, false);
    acc2 = __builtin_amdgcn_wmma_f32_16x16x32_bf16(false, a, false, b2, (short)0, acc2, false, false);
    acc3 = __builtin_amdgcn_wmma_f32_16x16x32_bf16(false, a, false, b3, (short)0, acc3, false, false);
  }

  float* orow = out + (size_t)(mBase + 8 * half) * DOUT + nBase + mrow;
#pragma unroll
  for (int r = 0; r < 8; ++r) {
    orow[(size_t)r * DOUT + 0]  = acc0[r];
    orow[(size_t)r * DOUT + 16] = acc1[r];
    orow[(size_t)r * DOUT + 32] = acc2[r];
    orow[(size_t)r * DOUT + 48] = acc3[r];
  }
}

// ---------------------------------------------------------------------------
extern "C" void kernel_launch(void* const* d_in, const int* in_sizes, int n_in,
                              void* d_out, int out_size, void* d_ws, size_t ws_size,
                              hipStream_t stream) {
  (void)in_sizes; (void)n_in; (void)out_size; (void)ws_size;
  const float* x   = (const float*)d_in[0];
  // d_in[1] routing_weights: unused (matches reference).
  const int*   idx = (const int*)d_in[2];
  const float* u   = (const float*)d_in[3];
  const float* svh = (const float*)d_in[4];
  float* out = (float*)d_out;

  // Workspace layout (~8.4 MB total):
  __bf16* svhT = (__bf16*)d_ws;                       // [DIN][EKN]   1 MB
  __bf16* utb  = svhT + (size_t)DIN * EKN;            // [EKN][DOUT]  1 MB
  float*  P    = (float*)(utb + (size_t)EKN * DOUT);  // [T][EKN]     4 MB
  __bf16* Wbf  = (__bf16*)(P + (size_t)T_TOK * EKN);  // [T][EKN]     2 MB

  prep_transpose<<<(DIN * EKN) / 256, 256, 0, stream>>>(u, svh, svhT, utb);
  gemm1_x_svhT<<<dim3(T_TOK / 128, EKN / 64), 256, 0, stream>>>(x, svhT, P);
  build_w<<<T_TOK, 256, 0, stream>>>(P, idx, Wbf);
  gemm2_w_ut<<<dim3(T_TOK / 128, DOUT / 64), 256, 0, stream>>>(Wbf, utb, out);
}
